// FlashinferNVFP4Dynamic_61040075211289
// MI455X (gfx1250) — matrix operand order repaired and checked by reading, not compiled, not run on hardware
//
#include <hip/hip_runtime.h>

typedef __attribute__((ext_vector_type(16))) _Float16 v16h;
typedef __attribute__((ext_vector_type(8)))  _Float16 v8h;
typedef __attribute__((ext_vector_type(8)))  float    v8f;

#define MDIM 4096
#define NDIM 4096
#define KDIM 4096
#define BM   128
#define BN   128
#define KB   64            // two 16x16x32 WMMA K-steps per staged tile
#define APAD 8             // As row stride 72 halves (144B) -> conflict-free b128 reads
#define BPAD 8             // Bs row stride 136 halves (272B)
#define ASZ  (BM * (KB + APAD))
#define BSZ  (KB * (BN + BPAD))

// ---------- NVFP4 rounding helpers ----------

// round-to-nearest-even onto e4m3 grid; v in [0, 448]
__device__ __forceinline__ float round_e4m3(float v) {
    if (!(v > 0.f)) return 0.f;
    if (v >= 0.015625f) {               // normal range (>= 2^-6): RNE to 3 mantissa bits
        union { float f; unsigned u; } c; c.f = v;
        unsigned lsb = (c.u >> 20) & 1u;
        c.u += 0x7FFFFu + lsb;          // round half to even at bit 20
        c.u &= 0xFFF00000u;
        return c.f;
    }
    return rintf(v * 512.f) * (1.f / 512.f);   // denormal quantum 2^-9
}

// nearest on e2m1 grid {0,.5,1,1.5,2,3,4,5,6}; ties -> lower (matches searchsorted 'left')
__device__ __forceinline__ float round_e2m1(float a) {
    if (a <= 0.25f) return 0.f;
    if (a <= 0.75f) return 0.5f;
    if (a <= 1.25f) return 1.f;
    if (a <= 1.75f) return 1.5f;
    if (a <= 2.5f)  return 2.f;
    if (a <= 3.5f)  return 3.f;
    if (a <= 4.5f)  return 4.f;
    if (a <= 5.5f)  return 5.f;
    return 6.f;
}

// ---------- phase 0: header init ----------
__global__ void init_hdr(unsigned* hdr) { hdr[0] = 0u; hdr[1] = 0u; }

// ---------- phase 1: global amax ----------
__global__ __launch_bounds__(256) void amax_kernel(const float* __restrict__ t, size_t n,
                                                   unsigned* __restrict__ dst) {
    float m = 0.f;
    for (size_t i = (size_t)blockIdx.x * blockDim.x + threadIdx.x; i < n;
         i += (size_t)gridDim.x * blockDim.x) {
        float v = fabsf(t[i]);
        if (v == v) m = fmaxf(m, v);    // nan_to_num
    }
    #pragma unroll
    for (int off = 16; off > 0; off >>= 1) m = fmaxf(m, __shfl_xor(m, off, 32));
    __shared__ float sm[8];
    int lane = threadIdx.x & 31, wid = threadIdx.x >> 5;
    if (lane == 0) sm[wid] = m;
    __syncthreads();
    if (threadIdx.x == 0) {
        float r = sm[0];
        #pragma unroll
        for (int i = 1; i < 8; i++) r = fmaxf(r, sm[i]);
        atomicMax(dst, __float_as_uint(r));   // non-negative: uint order == float order
    }
}

// ---------- phase 2: quant-dequant to exact f16 ----------
__global__ __launch_bounds__(256) void quant_kernel(const float* __restrict__ t,
                                                    _Float16* __restrict__ o,
                                                    const unsigned* __restrict__ hdr,
                                                    int which, size_t nblk) {
    size_t b = (size_t)blockIdx.x * blockDim.x + threadIdx.x;
    if (b >= nblk) return;
    float amax_g = fmaxf(__uint_as_float(hdr[which]), 1e-12f);
    float gsf = 2688.f / amax_g;                       // 448*6/amax
    const float4* p = (const float4*)(t + b * 16);
    float4 r0 = p[0], r1 = p[1], r2 = p[2], r3 = p[3];
    float v[16] = {r0.x, r0.y, r0.z, r0.w, r1.x, r1.y, r1.z, r1.w,
                   r2.x, r2.y, r2.z, r2.w, r3.x, r3.y, r3.z, r3.w};
    float am = 0.f;
    #pragma unroll
    for (int i = 0; i < 16; i++) am = fmaxf(am, fabsf(v[i]));
    float sf  = round_e4m3(fminf(am * gsf * (1.f / 6.f), 448.f));
    float inv = (sf > 0.f) ? (gsf / sf) : 0.f;
    union { _Float16 h[16]; uint4 u[2]; } outp;
    #pragma unroll
    for (int i = 0; i < 16; i++) {
        float s = v[i] * inv;
        float q = round_e2m1(fminf(fabsf(s), 6.f));
        q = (s < 0.f) ? -q : q;
        outp.h[i] = (_Float16)(q * sf);                // exact: <=6 significand bits
    }
    uint4* po = (uint4*)(o + b * 16);
    po[0] = outp.u[0];
    po[1] = outp.u[1];
}

// ---------- phase 3: WMMA GEMM  out = xh @ wh^T * alpha ----------
__global__ __launch_bounds__(256) void gemm_kernel(const _Float16* __restrict__ X,
                                                   const _Float16* __restrict__ W,
                                                   const unsigned* __restrict__ hdr,
                                                   float* __restrict__ out) {
    __shared__ _Float16 As[2 * ASZ];   // x tile, row-major [m][k], double buffered
    __shared__ _Float16 Bs[2 * BSZ];   // w tile transposed [k][n], double buffered

    const int tid  = threadIdx.x;
    const int lane = tid & 31;
    const int wid  = tid >> 5;                  // 8 waves
    const int wm   = (wid & 3) * 32;            // wave row offset in block tile (4 x M)
    const int wn   = (wid >> 2) * 64;           // wave col offset in block tile (2 x N)
    const int gm   = blockIdx.y * BM;
    const int gn   = blockIdx.x * BN;

    v8f zero = {};
    v8f acc[2][4];
    #pragma unroll
    for (int i = 0; i < 2; i++)
        #pragma unroll
        for (int j = 0; j < 4; j++) acc[i][j] = zero;

    const int mrow  = lane & 15;                // A frag: M = lane % 16
    const int kbase = (lane >> 4) * 8;          // A frag: K halves 0-7 / 8-15 per half-wave

    // A staging map: thread -> (row, 32-half chunk)
    const int arow = tid >> 1;
    const int acol = (tid & 1) * 32;
    // B staging map: thread -> (N pair, 8-wide K chunk), packed b32 transpose stores
    const int bnp  = tid >> 2;                  // 0..63 -> n = 2*bnp
    const int bk0  = (tid & 3) * 8;             // K chunk base (s adds 32)

    auto stage = [&](int buf, int kb) {
        // ---- A: 128x64 halves, straight copy, b128 in / b128 out ----
        const _Float16* xp = X + (size_t)(gm + arow) * KDIM + kb + acol;
        _Float16* ap = As + buf * ASZ + arow * (KB + APAD) + acol;
        #pragma unroll
        for (int c = 0; c < 4; c++)
            *(v8h*)(ap + c * 8) = *(const v8h*)(xp + c * 8);
        // ---- B: transpose [n][k] -> [k][n]; pack 2 N-rows into b32 LDS stores ----
        #pragma unroll
        for (int s = 0; s < 2; s++) {
            int k8 = bk0 + s * 32;
            const _Float16* wp = W + (size_t)(gn + 2 * bnp) * KDIM + kb + k8;
            v8h b0 = *(const v8h*)(wp);
            v8h b1 = *(const v8h*)(wp + KDIM);
            _Float16* bp = Bs + buf * BSZ + 2 * bnp;
            #pragma unroll
            for (int j = 0; j < 8; j++) {
                union { _Float16 h[2]; unsigned u; } pk;
                pk.h[0] = b0[j];
                pk.h[1] = b1[j];
                *(unsigned*)(bp + (k8 + j) * (BN + BPAD)) = pk.u;
            }
        }
        // ---- prefetch tile after next into L2/WGP$ (global_prefetch_b8) ----
        if (kb + KB < KDIM) {
            __builtin_prefetch(X + (size_t)(gm + arow) * KDIM + kb + KB + acol, 0, 1);
            __builtin_prefetch(W + (size_t)(gn + 2 * bnp) * KDIM + kb + KB + bk0, 0, 1);
        }
    };

    stage(0, 0);
    __syncthreads();

    const int ntiles = KDIM / KB;
    for (int kt = 0; kt < ntiles; kt++) {
        int cur = kt & 1;
        if (kt + 1 < ntiles) stage(cur ^ 1, (kt + 1) * KB);

        const _Float16* ab = As + cur * ASZ;
        const _Float16* bb = Bs + cur * BSZ;
        #pragma unroll
        for (int ks = 0; ks < 2; ks++) {        // two 32-wide WMMA K-steps per tile
            union V16 { v16h v; v8h h[2]; };
            V16 a[2], bf[4];
            #pragma unroll
            for (int i = 0; i < 2; i++) {
                const _Float16* ap = ab + (wm + i * 16 + mrow) * (KB + APAD) + ks * 32;
                a[i].h[0] = *(const v8h*)(ap + kbase);
                a[i].h[1] = *(const v8h*)(ap + kbase + 16);
            }
            #pragma unroll
            for (int j = 0; j < 4; j++) {
                const _Float16* bp = bb + (ks * 32 + kbase) * (BN + BPAD) + wn + j * 16 + mrow;
                #pragma unroll
                for (int s = 0; s < 8; s++) {
                    bf[j].v[s]     = bp[s * (BN + BPAD)];
                    bf[j].v[8 + s] = bp[(16 + s) * (BN + BPAD)];
                }
            }
            #pragma unroll
            for (int i = 0; i < 2; i++)
                #pragma unroll
                for (int j = 0; j < 4; j++)
                    acc[i][j] = __builtin_amdgcn_wmma_f32_16x16x32_f16(
                        false, a[i].v, false, bf[j].v, (short)0, acc[i][j], false, false);
        }
        __syncthreads();
    }

    // alpha = 1/(gsf_x*gsf_w) = amax_x*amax_w / 2688^2
    float ax = fmaxf(__uint_as_float(hdr[0]), 1e-12f);
    float aw = fmaxf(__uint_as_float(hdr[1]), 1e-12f);
    float alpha = (ax * aw) * (1.f / (2688.f * 2688.f));

    // C layout: VGPR r -> M = r + 8*(lane>=16), N = lane%16
    const int moff = (lane >> 4) * 8;
    const int ncol = lane & 15;
    #pragma unroll
    for (int i = 0; i < 2; i++)
        #pragma unroll
        for (int j = 0; j < 4; j++) {
            #pragma unroll
            for (int r = 0; r < 8; r++) {
                int m = gm + wm + i * 16 + moff + r;
                int n = gn + wn + j * 16 + ncol;
                out[(size_t)m * NDIM + n] = acc[i][j][r] * alpha;
            }
        }
}

extern "C" void kernel_launch(void* const* d_in, const int* in_sizes, int n_in,
                              void* d_out, int out_size, void* d_ws, size_t ws_size,
                              hipStream_t stream) {
    const float* x = (const float*)d_in[0];     // [M, K] f32
    const float* w = (const float*)d_in[1];     // [N, K] f32
    float* out = (float*)d_out;                 // [M, N] f32

    const size_t MK = (size_t)MDIM * KDIM;
    const size_t NK = (size_t)NDIM * KDIM;

    unsigned* hdr = (unsigned*)d_ws;                         // [0]=amax_x bits, [1]=amax_w bits
    _Float16* xh = (_Float16*)((char*)d_ws + 256);           // 32 MB
    _Float16* wh = xh + MK;                                  // 32 MB

    init_hdr<<<1, 1, 0, stream>>>(hdr);
    amax_kernel<<<1024, 256, 0, stream>>>(x, MK, hdr + 0);
    amax_kernel<<<1024, 256, 0, stream>>>(w, NK, hdr + 1);
    quant_kernel<<<(int)(MK / 16 / 256), 256, 0, stream>>>(x, xh, hdr, 0, MK / 16);
    quant_kernel<<<(int)(NK / 16 / 256), 256, 0, stream>>>(w, wh, hdr, 1, NK / 16);

    dim3 grid(NDIM / BN, MDIM / BM);
    gemm_kernel<<<grid, 256, 0, stream>>>(xh, wh, hdr, out);
}
